// RNN_44581760533079
// MI455X (gfx1250) — matrix operand order repaired and checked by reading, not compiled, not run on hardware
//
#include <hip/hip_runtime.h>
#include <hip/hip_bf16.h>
#include <stdint.h>

// Problem constants (match reference)
#define SEQ     512
#define BATCH   512
#define DIN     256
#define DH      256
#define KDIM    512     // DIN + DH
#define DOUT    2
#define PAD_IDX 1

// Kernel config
#define ROWS_PER_WG 16
#define NUM_WG      (BATCH / ROWS_PER_WG)   // 32 persistent workgroups
#define THREADS     256                      // 8 wave32 waves

// LDS layout (fragment-native, so every WMMA operand = two ds_load_b128):
//  Wf: B-fragments of W_i2h, [nt=16][ks=16] regions of 256 dwords:
//      chunk0 (dw 0..127):  lane*4 + j      (VGPRs j=0..3)
//      chunk1 (dw 128..255): lane*4 + (j-4) (VGPRs j=4..7)
//  Af: A-fragments of combined [x|h], [ks=16] regions of 256 dwords, same scheme
//  Hlds: fp32 hidden slice [16][258] (even stride -> aligned float2 reads)
#define WF_DW  (16 * 16 * 256)   // 65536 dwords = 256 KB
#define AF_DW  (16 * 256)        //  4096 dwords =  16 KB
#define H_S    258
#define H_DW   (16 * H_S)        //  4128 dwords

#define WF_BYTES (WF_DW * 4)
#define AF_BYTES (AF_DW * 4)
#define H_BYTES  (H_DW * 4)
#define SMEM_BYTES (WF_BYTES + AF_BYTES + H_BYTES)   // 295,040 B < 320 KB

typedef __attribute__((ext_vector_type(16))) __bf16 v16bf;
typedef __attribute__((ext_vector_type(8)))  float  v8f;

union Frag { v16bf v; struct { uint4 lo, hi; } q; };

// round-to-nearest-even f32 -> bf16, two packed into a dword
__device__ __forceinline__ uint32_t pack_bf16(float lo, float hi) {
  uint32_t a = __float_as_uint(lo);
  uint32_t b = __float_as_uint(hi);
  a = (a + 0x7FFFu + ((a >> 16) & 1u)) >> 16;
  b = (b + 0x7FFFu + ((b >> 16) & 1u)) >> 16;
  return (a & 0xFFFFu) | (b << 16);
}

// A-matrix (16-bit 16x32) fragment slot for element (row m, global K-pair pjFull):
//   lanes 0-15: M=m, K-pairs {0..3}->j0..3, {8..11}->j4..7
//   lanes 16-31: K-pairs {4..7}->j0..3, {12..15}->j4..7
__device__ __forceinline__ int a_slot(int m, int pjFull) {
  int ks    = pjFull >> 4;
  int lp    = pjFull & 15;
  int lane  = m + ((lp & 4) ? 16 : 0);
  int chunk = (lp & 8) ? 128 : 0;
  return ks * 256 + chunk + lane * 4 + (lp & 3);
}

extern "C" __global__ __launch_bounds__(THREADS, 1)
void rnn_persist_wmma(const int*   __restrict__ text,     // [SEQ][BATCH]
                      const int*   __restrict__ ubnd,     // [SEQ]
                      const float* __restrict__ hidden0,  // [BATCH][DH]
                      const float* __restrict__ emb,      // [VOCAB][DIN]
                      const float* __restrict__ Wih,      // [KDIM][DH]
                      const float* __restrict__ bih,      // [DH]
                      const float* __restrict__ Wio,      // [KDIM][DOUT]
                      const float* __restrict__ bio,      // [DOUT]
                      float*       __restrict__ out)      // [BATCH][DOUT]
{
  extern __shared__ unsigned char smem_raw[];
  uint32_t* Wf   = (uint32_t*)(smem_raw);
  uint32_t* Af   = (uint32_t*)(smem_raw + WF_BYTES);
  float*    Hlds = (float*)   (smem_raw + WF_BYTES + AF_BYTES);

  const int tid  = threadIdx.x;
  const int lane = tid & 31;
  const int wv   = tid >> 5;          // wave 0..7
  const int b0   = blockIdx.x * ROWS_PER_WG;
  const int hi16 = (lane >> 4) & 1;
  const int l15  = lane & 15;

  // ---- one-time init: swizzle W_i2h into bf16 B-fragments, load hidden0 ----
  // B-matrix (16-bit 32x16): lane L -> col N=(L&15); lanes 0-15 K-pairs
  // {0..3}->j0..3, {8..11}->j4..7; lanes 16-31 K-pairs {4..7}->j0..3, {12..15}->j4..7.
  #pragma unroll 4
  for (int i = 0; i < 256; ++i) {
    int q  = tid + i * THREADS;       // 0..65535
    int p  = q >> 8;                  // global K-pair 0..255 (K = 2p, 2p+1)
    int n  = q & 255;                 // output column
    int nt = n >> 4, nl = n & 15;
    int ks = p >> 4, tp = p & 15;
    int j  = (tp & 3) | ((tp >> 1) & 4), khalf = (tp >> 2) & 1;
    int ln = nl + 16 * khalf;
    int dst = (nt * 16 + ks) * 256 + ((j & 4) ? 128 : 0) + ln * 4 + (j & 3);
    Wf[dst] = pack_bf16(Wih[(2 * p) * DH + n], Wih[(2 * p + 1) * DH + n]);
  }
  #pragma unroll
  for (int i = 0; i < 16; ++i) {
    int q = tid + i * THREADS;        // 0..4095
    int m = q >> 8, n = q & 255;
    Hlds[m * H_S + n] = hidden0[(b0 + m) * DH + n];
  }
  const int   n0a   = wv * 32;        // this wave's two output-column tiles
  const int   n0b   = n0a + 16;
  const int   ntA   = wv * 2;         // fragment-region tile ids
  const int   ntB   = wv * 2 + 1;
  const float biasA = bih[n0a + l15];
  const float biasB = bih[n0b + l15];
  __syncthreads();

  // ================= sequential recurrence: t = 0 .. 510 =================
  for (int t = 0; t < SEQ - 1; ++t) {
    // ---- stage combined = [x_t | h] straight into A-fragment layout ----
    #pragma unroll
    for (int i = 0; i < 8; ++i) {                 // x part: K pairs 0..127
      int q  = tid + i * THREADS;                 // 0..2047
      int m  = q >> 7;
      int pj = q & 127;
      int idx = text[t * BATCH + b0 + m];
      float lo = 0.f, hi = 0.f;
      if (idx != PAD_IDX) {                       // padding_idx row is zero
        const float* er = emb + (size_t)idx * DIN + 2 * pj;
        lo = er[0]; hi = er[1];
      }
      Af[a_slot(m, pj)] = pack_bf16(lo, hi);
    }
    #pragma unroll
    for (int i = 0; i < 8; ++i) {                 // hidden part: K pairs 128..255
      int q  = tid + i * THREADS;
      int m  = q >> 7;
      int pj = q & 127;
      float2 h2 = *(const float2*)(Hlds + m * H_S + 2 * pj);   // H_S even -> aligned
      Af[a_slot(m, 128 + pj)] = pack_bf16(h2.x, h2.y);
    }
    // prefetch next step's embedding rows toward cache (global_prefetch_b8)
    {
      int m = tid >> 4, seg = tid & 15;
      int idxn = text[(t + 1) * BATCH + b0 + m];
      if (idxn != PAD_IDX)
        __builtin_prefetch(emb + (size_t)idxn * DIN + seg * 16, 0, 1);
    }
    __syncthreads();

    // ---- WMMA: one 16x16 M-tile x two N-tiles per wave, K = 512 ----
    // 2 partial accumulators per tile -> 4-deep WMMA round-robin hides the
    // 5-slot bf16 WMMA->WMMA RAW hazard window.
    v8f acc0[2] = {{}, {}};
    v8f acc1[2] = {{}, {}};
    #pragma unroll
    for (int ks = 0; ks < 16; ++ks) {
      Frag a, bA, bB;
      const uint32_t* ab = Af + ks * 256;
      a.q.lo  = *(const uint4*)(ab + lane * 4);
      a.q.hi  = *(const uint4*)(ab + 128 + lane * 4);
      const uint32_t* wa = Wf + (ntA * 16 + ks) * 256;
      bA.q.lo = *(const uint4*)(wa + lane * 4);
      bA.q.hi = *(const uint4*)(wa + 128 + lane * 4);
      const uint32_t* wb = Wf + (ntB * 16 + ks) * 256;
      bB.q.lo = *(const uint4*)(wb + lane * 4);
      bB.q.hi = *(const uint4*)(wb + 128 + lane * 4);
      int s = ks & 1;
      acc0[s] = __builtin_amdgcn_wmma_f32_16x16x32_bf16(false, a.v, false, bA.v,
                                                        (short)0, acc0[s], false, false);
      acc1[s] = __builtin_amdgcn_wmma_f32_16x16x32_bf16(false, a.v, false, bB.v,
                                                        (short)0, acc1[s], false, false);
    }

    // ---- bias + update-bound mask, write hidden back in fp32 ----
    // D layout: VGPR r -> (M = r + 8*hi16, N = n0 + (lane&15))
    const int ubt = ubnd[t];
    #pragma unroll
    for (int r = 0; r < 8; ++r) {
      int   m   = r + (hi16 ? 8 : 0);
      bool  upd = (b0 + m) < ubt;
      float n0v = acc0[0][r] + acc0[1][r] + biasA;
      float n1v = acc1[0][r] + acc1[1][r] + biasB;
      float o0  = Hlds[m * H_S + n0a + l15];
      float o1  = Hlds[m * H_S + n0b + l15];
      Hlds[m * H_S + n0a + l15] = upd ? n0v : o0;
      Hlds[m * H_S + n0b + l15] = upd ? n1v : o1;
    }
    __syncthreads();
  }

  // ================= final step: only outputs[-1] matters =================
  // output[b] = [x_511 | h] @ W_i2o + b_i2o, with h = hidden entering t=511
  // (the t=511 hidden update is never observed -> skip it entirely)
  if (tid < ROWS_PER_WG * DOUT) {
    int m = tid >> 1, o = tid & 1;
    int idx = text[(SEQ - 1) * BATCH + b0 + m];
    float acc = bio[o];
    if (idx != PAD_IDX) {
      const float* er = emb + (size_t)idx * DIN;
      #pragma unroll 4
      for (int k = 0; k < DIN; ++k) acc += er[k] * Wio[k * DOUT + o];
    }
    #pragma unroll 4
    for (int k = 0; k < DH; ++k) acc += Hlds[m * H_S + k] * Wio[(DIN + k) * DOUT + o];
    out[(b0 + m) * DOUT + o] = acc;
  }
}

extern "C" void kernel_launch(void* const* d_in, const int* in_sizes, int n_in,
                              void* d_out, int out_size, void* d_ws, size_t ws_size,
                              hipStream_t stream) {
  (void)in_sizes; (void)n_in; (void)out_size; (void)d_ws; (void)ws_size;
  const int*   text = (const int*)  d_in[0];
  const int*   ub   = (const int*)  d_in[1];
  const float* h0   = (const float*)d_in[2];
  const float* emb  = (const float*)d_in[3];
  const float* Wih  = (const float*)d_in[4];
  const float* bih  = (const float*)d_in[5];
  const float* Wio  = (const float*)d_in[6];
  const float* bio  = (const float*)d_in[7];
  float* out = (float*)d_out;

  // CDNA5 allows up to 320KB LDS per workgroup; request our ~288KB.
  (void)hipFuncSetAttribute((const void*)rnn_persist_wmma,
                            hipFuncAttributeMaxDynamicSharedMemorySize,
                            (int)SMEM_BYTES);

  rnn_persist_wmma<<<NUM_WG, THREADS, SMEM_BYTES, stream>>>(
      text, ub, h0, emb, Wih, bih, Wio, bio, out);
}
